// RGCN_26053271618133
// MI455X (gfx1250) — matrix operand order repaired and checked by reading, not compiled, not run on hardware
//
#include <hip/hip_runtime.h>
#include <stdint.h>

#define NN 100000
#define EE 1600000
#define RN 8
#define BN 4

typedef __bf16 bf16;
typedef __attribute__((ext_vector_type(16))) __bf16 v16bf;
typedef __attribute__((ext_vector_type(8)))  __bf16 v8bf;
typedef __attribute__((ext_vector_type(8)))  float  v8f;

static __device__ __forceinline__ v8f wmma_bf16(v16bf a, v16bf b, v8f c) {
  // (neg_a, A, neg_b, B, c_mod, C, reuse_a, reuse_b)
  return __builtin_amdgcn_wmma_f32_16x16x32_bf16(false, a, false, b, (short)0, c, false, false);
}

// ---------------- utility fills ----------------
__global__ void k_fill_u32(unsigned* __restrict__ p, unsigned v, long n) {
  long i = (long)blockIdx.x * blockDim.x + threadIdx.x;
  if (i < n) p[i] = v;
}
__global__ void k_fill_f32(float* __restrict__ p, float v, long n) {
  long i = (long)blockIdx.x * blockDim.x + threadIdx.x;
  if (i < n) p[i] = v;
}

// ---------------- (dst,rel) degree count + per-relation histogram ----------------
__global__ void k_count(const int* __restrict__ ei, const int* __restrict__ et,
                        unsigned* __restrict__ cnt, unsigned* __restrict__ relCnt) {
  int e = blockIdx.x * blockDim.x + threadIdx.x;
  if (e >= EE) return;
  int d = ei[EE + e];
  int r = et[e];
  atomicAdd(&cnt[(long)d * RN + r], 1u);
  atomicAdd(&relCnt[r], 1u);
}

// 16-aligned exclusive prefix over relation counts (single thread; tiny)
__global__ void k_prefix(const unsigned* __restrict__ relCnt,
                         unsigned* __restrict__ aoff, unsigned* __restrict__ cur) {
  if (blockIdx.x == 0 && threadIdx.x == 0) {
    unsigned o = 0;
    for (int r = 0; r < RN; r++) {
      aoff[r] = o; cur[r] = 0;
      o += (relCnt[r] + 15u) & ~15u;
    }
    aoff[RN] = o;
  }
}

// scatter edges into relation bins; sentinel slots keep norm==0
__global__ void k_bin(const int* __restrict__ ei, const int* __restrict__ et,
                      const unsigned* __restrict__ cnt, const unsigned* __restrict__ aoff,
                      unsigned* __restrict__ cur,
                      unsigned* __restrict__ sSrc, unsigned* __restrict__ sDst,
                      float* __restrict__ sNorm) {
  int e = blockIdx.x * blockDim.x + threadIdx.x;
  if (e >= EE) return;
  int s = ei[e], d = ei[EE + e], r = et[e];
  unsigned pos = atomicAdd(&cur[r], 1u);
  unsigned idx = aoff[r] + pos;
  sSrc[idx] = (unsigned)s;
  sDst[idx] = (unsigned)d;
  unsigned c = cnt[(long)d * RN + r];
  sNorm[idx] = 1.0f / (float)(c < 1u ? 1u : c);
}

// ---------------- build W_r = sum_b comp[r,b]*bases[b], pre-swizzled WMMA-B layout, bf16 ----------
// B-layout (32x16 bf16): lane l -> column n=l&15, k = kt*32 + (l>>4)*16 + j (j=0..15 contiguous)
__global__ void k_buildW(const float* __restrict__ bases, const float* __restrict__ comp,
                         bf16* __restrict__ w, int din, int dout, int nT, int kT) {
  long i = (long)blockIdx.x * blockDim.x + threadIdx.x;
  long total = (long)RN * nT * kT * 512;
  if (i >= total) return;
  int j = (int)(i & 15);
  int l = (int)((i >> 4) & 31);
  long q = i >> 9;
  int kt = (int)(q % kT); q /= kT;
  int t  = (int)(q % nT);
  int r  = (int)(q / nT);
  int n = t * 16 + (l & 15);
  int k = kt * 32 + ((l >> 4) << 4) + j;
  float v = 0.f;
  if (n < dout) {
#pragma unroll
    for (int b = 0; b < BN; b++)
      v += comp[r * BN + b] * bases[((long)b * din + k) * dout + n];
  }
  w[i] = (bf16)v;
}

__global__ void k_buildRoot(const float* __restrict__ root, bf16* __restrict__ w,
                            int din, int dout, int nT, int kT) {
  long i = (long)blockIdx.x * blockDim.x + threadIdx.x;
  long total = (long)nT * kT * 512;
  if (i >= total) return;
  int j = (int)(i & 15);
  int l = (int)((i >> 4) & 31);
  long q = i >> 9;
  int kt = (int)(q % kT);
  int t  = (int)(q / kT);
  int n = t * 16 + (l & 15);
  int k = kt * 32 + ((l >> 4) << 4) + j;
  float v = (n < dout) ? root[(long)k * dout + n] : 0.f;
  w[i] = (bf16)v;
}

__global__ void k_cvt_bf16(const float* __restrict__ x, bf16* __restrict__ y, long n) {
  long i = (long)blockIdx.x * blockDim.x + threadIdx.x;
  if (i < n) y[i] = (bf16)x[i];
}

// load A-operand (16x32 bf16, rows = 16 gathered node rows) for one K-half pair
// A-layout: lane l -> row m=l&15; k = hh*16..hh*16+15 (contiguous) per 32-K tile, hh=l>>4
static __device__ __forceinline__ void load_a64(const bf16* __restrict__ row,
                                                int hh, v16bf& a0, v16bf& a1) {
  const v8bf* hv = (const v8bf*)row;
  v8bf a0lo = hv[2 * hh],     a0hi = hv[2 * hh + 1];
  v8bf a1lo = hv[2 * hh + 4], a1hi = hv[2 * hh + 5];
#pragma unroll
  for (int j = 0; j < 8; j++) {
    a0[j] = a0lo[j]; a0[8 + j] = a0hi[j];
    a1[j] = a1lo[j]; a1[8 + j] = a1hi[j];
  }
}

// ---------------- edge-message kernel: one wave = one 16-edge tile (single relation) ----------
template <int DOUT>
__global__ void k_edge(const unsigned* __restrict__ sSrc, const unsigned* __restrict__ sDst,
                       const float* __restrict__ sNorm, const unsigned* __restrict__ aoff,
                       const bf16* __restrict__ hbf, const bf16* __restrict__ wW,
                       float* __restrict__ agg) {
  const int nT = (DOUT + 15) / 16;
  int wave = threadIdx.x >> 5, lane = threadIdx.x & 31;
  long tile = (long)blockIdx.x * (blockDim.x >> 5) + wave;
  unsigned p0 = (unsigned)(tile * 16);
  if (p0 >= aoff[RN]) return;          // wave-uniform: EXEC stays full for WMMA
  // warm the scatter metadata lines while the WMMA pipeline runs
  __builtin_prefetch(&sDst[p0], 0, 1);
  __builtin_prefetch(&sNorm[p0], 0, 1);
  int r = 0;
#pragma unroll
  for (int i = 1; i < RN; i++) r += (p0 >= aoff[i]) ? 1 : 0;

  int m = lane & 15, hh = lane >> 4;
  unsigned src = sSrc[p0 + (unsigned)m];

  v16bf a0, a1;
  load_a64(hbf + (long)src * 64, hh, a0, a1);

  const v16bf* wv = (const v16bf*)wW;
  v8f acc[nT];
#pragma unroll
  for (int t = 0; t < nT; t++) {
    v8f c = {};
    v16bf b0 = wv[(long)((r * nT + t) * 2 + 0) * 32 + lane];
    v16bf b1 = wv[(long)((r * nT + t) * 2 + 1) * 32 + lane];
    c = wmma_bf16(a0, b0, c);
    c = wmma_bf16(a1, b1, c);
    acc[t] = c;
  }

  // scatter: C layout -> VGPR j holds row M = j + hh*8, col N = m.
  // Only 2 distinct metadata addresses per wave per j (broadcast-friendly loads).
#pragma unroll
  for (int j = 0; j < 8; j++) {
    int M = j + (hh << 3);
    unsigned d = sDst[p0 + (unsigned)M];
    float   nm = sNorm[p0 + (unsigned)M];   // 0 => padding sentinel
    if (nm != 0.f) {
      float* base = agg + (long)d * DOUT + m;
#pragma unroll
      for (int t = 0; t < nT; t++) {
        int n = t * 16 + m;
        if (DOUT >= 16 || n < DOUT)
          atomicAdd(base + t * 16, acc[t][j] * nm);   // immediate-offset atomics
      }
    }
  }
}

// ---------------- finalize: out = agg + h@root + bias (+ReLU); writes bf16 activations ----------
template <int DOUT, bool RELU, bool WF, bool WBF>
__global__ void k_finalize(const bf16* __restrict__ hbf, const bf16* __restrict__ wRoot,
                           const float* __restrict__ agg, const float* __restrict__ bias,
                           float* __restrict__ outF, bf16* __restrict__ outB) {
  const int nT = (DOUT + 15) / 16;
  int wave = threadIdx.x >> 5, lane = threadIdx.x & 31;
  long tile = (long)blockIdx.x * (blockDim.x >> 5) + wave;
  long i0 = tile * 16;
  if (i0 >= NN) return;                // N is a multiple of 16
  int m = lane & 15, hh = lane >> 4;
  long row = i0 + m;

  v16bf a0, a1;
  load_a64(hbf + row * 64, hh, a0, a1);

  const v16bf* wv = (const v16bf*)wRoot;
  v8f acc[nT];
#pragma unroll
  for (int t = 0; t < nT; t++) {
    v8f c = {};
    v16bf b0 = wv[(long)(t * 2 + 0) * 32 + lane];
    v16bf b1 = wv[(long)(t * 2 + 1) * 32 + lane];
    c = wmma_bf16(a0, b0, c);
    c = wmma_bf16(a1, b1, c);
    acc[t] = c;
  }

  float bv[nT];
#pragma unroll
  for (int t = 0; t < nT; t++) {
    int n = t * 16 + m;
    bv[t] = (DOUT >= 16 || n < DOUT) ? bias[n] : 0.f;
  }

#pragma unroll
  for (int j = 0; j < 8; j++) {
    int M = j + (hh << 3);
    long rr = i0 + M;
    long base = rr * DOUT + m;
#pragma unroll
    for (int t = 0; t < nT; t++) {
      int n = t * 16 + m;
      if (DOUT >= 16 || n < DOUT) {
        float v = acc[t][j] + agg[base + t * 16] + bv[t];
        if (RELU) v = v > 0.f ? v : 0.f;
        if (WF)  outF[base + t * 16] = v;
        if (WBF) outB[base + t * 16] = (bf16)v;   // in-place activation update (row-exclusive)
      }
    }
  }
}

// ---------------- host driver ----------------
extern "C" void kernel_launch(void* const* d_in, const int* in_sizes, int n_in,
                              void* d_out, int out_size, void* d_ws, size_t ws_size,
                              hipStream_t stream) {
  (void)in_sizes; (void)n_in; (void)out_size; (void)ws_size;
  const float* x  = (const float*)d_in[0];
  const int*   ei = (const int*)d_in[1];
  const int*   et = (const int*)d_in[2];
  const float* bases[3] = {(const float*)d_in[3], (const float*)d_in[7],  (const float*)d_in[11]};
  const float* comp[3]  = {(const float*)d_in[4], (const float*)d_in[8],  (const float*)d_in[12]};
  const float* root[3]  = {(const float*)d_in[5], (const float*)d_in[9],  (const float*)d_in[13]};
  const float* bias[3]  = {(const float*)d_in[6], (const float*)d_in[10], (const float*)d_in[14]};

  char* w = (char*)d_ws;
  auto alloc = [&](size_t bytes) -> char* {
    char* p = w; w += (bytes + 255) & ~(size_t)255; return p;
  };
  const long CAP = (long)EE + 16 * RN;
  unsigned* cnt    = (unsigned*)alloc((size_t)NN * RN * 4);
  unsigned* relCnt = (unsigned*)alloc(8 * 4);
  unsigned* aoff   = (unsigned*)alloc(9 * 4);
  unsigned* cur    = (unsigned*)alloc(8 * 4);
  unsigned* sSrc   = (unsigned*)alloc((size_t)CAP * 4);
  unsigned* sDst   = (unsigned*)alloc((size_t)CAP * 4);
  float*    sNorm  = (float*)alloc((size_t)CAP * 4);
  bf16*     wW     = (bf16*)alloc((size_t)RN * 4 * 2 * 512 * 2);
  bf16*     wRoot  = (bf16*)alloc((size_t)4 * 2 * 512 * 2);
  bf16*     hbf    = (bf16*)alloc((size_t)NN * 64 * 2);
  float*    agg    = (float*)alloc((size_t)NN * 64 * 4);

  auto cdiv = [](long a, long b) { return (unsigned)((a + b - 1) / b); };

  // one-time graph preprocessing (layer-invariant)
  k_fill_u32<<<cdiv((long)NN * RN, 256), 256, 0, stream>>>(cnt, 0u, (long)NN * RN);
  k_fill_u32<<<1, 32, 0, stream>>>(relCnt, 0u, 8);
  k_count<<<cdiv(EE, 256), 256, 0, stream>>>(ei, et, cnt, relCnt);
  k_prefix<<<1, 1, 0, stream>>>(relCnt, aoff, cur);
  k_fill_u32<<<cdiv(CAP, 256), 256, 0, stream>>>(sSrc, 0u, CAP);
  k_fill_u32<<<cdiv(CAP, 256), 256, 0, stream>>>(sDst, 0u, CAP);
  k_fill_f32<<<cdiv(CAP, 256), 256, 0, stream>>>(sNorm, 0.f, CAP);
  k_bin<<<cdiv(EE, 256), 256, 0, stream>>>(ei, et, cnt, aoff, cur, sSrc, sDst, sNorm);
  k_cvt_bf16<<<cdiv((long)NN * 64, 256), 256, 0, stream>>>(x, hbf, (long)NN * 64);

  const unsigned edgeBlocks = cdiv(EE / 16 + RN, 8);   // 8 waves/block, 1 tile/wave
  const unsigned nodeBlocks = cdiv((long)NN / 16, 8);

  // ---- layer 0 (64 -> 64, ReLU) ----
  k_buildW<<<cdiv((long)RN * 4 * 2 * 512, 256), 256, 0, stream>>>(bases[0], comp[0], wW, 64, 64, 4, 2);
  k_buildRoot<<<cdiv((long)4 * 2 * 512, 256), 256, 0, stream>>>(root[0], wRoot, 64, 64, 4, 2);
  k_fill_f32<<<cdiv((long)NN * 64, 256), 256, 0, stream>>>(agg, 0.f, (long)NN * 64);
  k_edge<64><<<edgeBlocks, 256, 0, stream>>>(sSrc, sDst, sNorm, aoff, hbf, wW, agg);
  k_finalize<64, true, false, true><<<nodeBlocks, 256, 0, stream>>>(hbf, wRoot, agg, bias[0], nullptr, hbf);

  // ---- layer 1 (64 -> 64, ReLU) ----
  k_buildW<<<cdiv((long)RN * 4 * 2 * 512, 256), 256, 0, stream>>>(bases[1], comp[1], wW, 64, 64, 4, 2);
  k_buildRoot<<<cdiv((long)4 * 2 * 512, 256), 256, 0, stream>>>(root[1], wRoot, 64, 64, 4, 2);
  k_fill_f32<<<cdiv((long)NN * 64, 256), 256, 0, stream>>>(agg, 0.f, (long)NN * 64);
  k_edge<64><<<edgeBlocks, 256, 0, stream>>>(sSrc, sDst, sNorm, aoff, hbf, wW, agg);
  k_finalize<64, true, false, true><<<nodeBlocks, 256, 0, stream>>>(hbf, wRoot, agg, bias[1], nullptr, hbf);

  // ---- layer 2 (64 -> 2, fp32 output) ----
  k_buildW<<<cdiv((long)RN * 1 * 2 * 512, 256), 256, 0, stream>>>(bases[2], comp[2], wW, 64, 2, 1, 2);
  k_buildRoot<<<cdiv((long)1 * 2 * 512, 256), 256, 0, stream>>>(root[2], wRoot, 64, 2, 1, 2);
  k_fill_f32<<<cdiv((long)NN * 2, 256), 256, 0, stream>>>(agg, 0.f, (long)NN * 2);
  k_edge<2><<<edgeBlocks, 256, 0, stream>>>(sSrc, sDst, sNorm, aoff, hbf, wW, agg);
  k_finalize<2, false, true, false><<<nodeBlocks, 256, 0, stream>>>(hbf, wRoot, agg, bias[2], (float*)d_out, nullptr);
}
